// GAT_3058016715241
// MI455X (gfx1250) — matrix operand order repaired and checked by reading, not compiled, not run on hardware
//
#include <hip/hip_runtime.h>
#include <hip/hip_bf16.h>
#include <math.h>

// ---------------- problem constants (from reference) ----------------
#define NNODES   40000
#define NEDGES   640000
#define ETOT     (NEDGES + NNODES)   // edges + self loops
#define NGRAPH   64
#define NEG_SLOPE 0.2f

typedef __attribute__((ext_vector_type(16))) _Float16 v16h;
typedef __attribute__((ext_vector_type(8)))  float    v8f;

// =====================================================================
// WMMA GEMM: C[M,Nc] = A[M,K] @ B[K,Nc], fp32 in/out, f16 WMMA compute.
// One wave computes a 16x32 output slab (two adjacent 16x16 n-tiles):
// the A fragment is loaded/converted once and feeds two v_wmma ops.
// M must be a multiple of 16 (40000 = 2500*16), Nc a multiple of 32.
// A/B lanes use contiguous k = hi*16 + j in both (not ISA table order); f32 C/D per CDNA5 ISA 7.12.2.
// =====================================================================
__global__ void gemm_wmma_f16(const float* __restrict__ A,
                              const float* __restrict__ B,
                              float* __restrict__ C,
                              int K, int Nc) {
    const int wave   = threadIdx.x >> 5;
    const int lane   = threadIdx.x & 31;
    const int mTile  = blockIdx.x;
    const int nBase  = (blockIdx.y * 8 + wave) * 32;   // 2 n-tiles per wave
    if (nBase >= Nc) return;                           // wave-uniform exit

    const int m16 = lane & 15;                 // A row / B-D column within tile
    const int hi  = lane >> 4;                 // half-wave selector
    const float* arow = A + (size_t)(mTile * 16 + m16) * K;

    v8f acc0 = {}, acc1 = {};
    for (int kt = 0; kt < K; kt += 32) {
        // ---- A tile 16x32: lane holds row m16; elem j -> K=hi*16+j ----
        v16h a;
        #pragma unroll
        for (int j = 0; j < 16; ++j) {
            int k = hi * 16 + j;
            a[j] = (_Float16)arow[kt + k];
        }
        // ---- two B tiles 32x16: lane holds column m16; elem j -> K=hi*16+j ----
        const float* bbase = B + (size_t)(kt + hi * 16) * Nc + nBase + m16;
        v16h b0, b1;
        #pragma unroll
        for (int j = 0; j < 16; ++j) {
            b0[j] = (_Float16)bbase[(size_t)j * Nc];
            b1[j] = (_Float16)bbase[(size_t)j * Nc + 16];
        }
        acc0 = __builtin_amdgcn_wmma_f32_16x16x32_f16(
                   false, a, false, b0, (short)0, acc0, false, false);
        acc1 = __builtin_amdgcn_wmma_f32_16x16x32_f16(
                   false, a, false, b1, (short)0, acc1, false, false);
    }
    // ---- C/D layout: VGPR r holds row (r + hi*8), col m16 ----
    float* crow = C + (size_t)(mTile * 16 + hi * 8) * Nc + nBase + m16;
    #pragma unroll
    for (int r = 0; r < 8; ++r) {
        crow[(size_t)r * Nc]      = acc0[r];
        crow[(size_t)r * Nc + 16] = acc1[r];
    }
}

// =====================================================================
// alpha_src / alpha_dst per (node, head): 64-wide dot products
// =====================================================================
__global__ void alpha_kernel(const float* __restrict__ h,
                             const float* __restrict__ a_src,
                             const float* __restrict__ a_dst,
                             float* __restrict__ alphaS,
                             float* __restrict__ alphaD, int H) {
    int i = blockIdx.x * blockDim.x + threadIdx.x;   // over N*H
    if (i >= NNODES * H) return;
    int n = i / H, hd = i % H;
    const float* hp = h + (size_t)n * H * 64 + hd * 64;
    const float* as = a_src + hd * 64;
    const float* ad = a_dst + hd * 64;
    float s = 0.f, d = 0.f;
    #pragma unroll 4
    for (int c = 0; c < 64; ++c) { float v = hp[c]; s += v * as[c]; d += v * ad[c]; }
    alphaS[i] = s; alphaD[i] = d;
}

// emax <- -inf, denom <- 0
__global__ void init_node_kernel(float* __restrict__ emax,
                                 float* __restrict__ denom, int count) {
    int i = blockIdx.x * blockDim.x + threadIdx.x;
    if (i < count) { emax[i] = -INFINITY; denom[i] = 0.f; }
}

// order-preserving float atomic max (lowers to global_atomic_max_i32 / min_u32)
__device__ __forceinline__ void atomicMaxF(float* addr, float v) {
    if (v >= 0.f) atomicMax((int*)addr, __float_as_int(v));
    else          atomicMin((unsigned int*)addr, __float_as_uint(v));
}

__device__ __forceinline__ void edge_ends(const int* __restrict__ ei, int e,
                                          int& src, int& dst) {
    if (e < NEDGES) { src = ei[e]; dst = ei[NEDGES + e]; }
    else            { src = dst = e - NEDGES; }          // self loop
}

// e = leaky_relu(aS[src]+aD[dst]); segment-max into emax[dst]
__global__ void edge_score_kernel(const int* __restrict__ ei,
                                  const float* __restrict__ alphaS,
                                  const float* __restrict__ alphaD,
                                  float* __restrict__ eBuf,
                                  float* __restrict__ emax, int H) {
    int e = blockIdx.x * blockDim.x + threadIdx.x;
    if (e >= ETOT) return;
    int src, dst; edge_ends(ei, e, src, dst);
    for (int hd = 0; hd < H; ++hd) {
        float v = alphaS[src * H + hd] + alphaD[dst * H + hd];
        v = (v > 0.f) ? v : NEG_SLOPE * v;
        eBuf[(size_t)e * H + hd] = v;
        atomicMaxF(&emax[dst * H + hd], v);
    }
}

// ex = exp(e - emax[dst]); segment-sum into denom[dst]; ex overwrites eBuf
__global__ void edge_exp_kernel(const int* __restrict__ ei,
                                float* __restrict__ eBuf,
                                const float* __restrict__ emax,
                                float* __restrict__ denom, int H) {
    int e = blockIdx.x * blockDim.x + threadIdx.x;
    if (e >= ETOT) return;
    int src, dst; edge_ends(ei, e, src, dst);
    for (int hd = 0; hd < H; ++hd) {
        float ex = __expf(eBuf[(size_t)e * H + hd] - emax[dst * H + hd]);
        eBuf[(size_t)e * H + hd] = ex;
        atomicAdd(&denom[dst * H + hd], ex);
    }
}

// accum[dst,h,c] += ex[e,h] * h[src,h,c]   (blockDim = H*64, 4 edges/block)
__global__ void edge_agg_kernel(const int* __restrict__ ei,
                                const float* __restrict__ eBuf,
                                const float* __restrict__ hbuf,
                                float* __restrict__ accum, int H) {
    const int t  = threadIdx.x;            // 0 .. H*64-1
    const int hd = t >> 6;
    const int HC = H << 6;
    const int base = blockIdx.x * 4;
    for (int k = 0; k < 4; ++k) {
        int e = base + k;
        if (e >= ETOT) return;
        int src, dst; edge_ends(ei, e, src, dst);
        float ex = eBuf[(size_t)e * H + hd];
        float v  = ex * hbuf[(size_t)src * HC + t];
        atomicAdd(&accum[(size_t)dst * HC + t], v);
    }
}

// out = accum/denom + bias, optional relu (in-place safe)
__global__ void finalize_kernel(const float* __restrict__ accum,
                                const float* __restrict__ denom,
                                const float* __restrict__ bias,
                                float* __restrict__ out, int H, int relu) {
    size_t total = (size_t)NNODES * H * 64;
    size_t i = (size_t)blockIdx.x * blockDim.x + threadIdx.x;
    if (i >= total) return;
    int HC = H << 6;
    int n  = (int)(i / HC);
    int t  = (int)(i % HC);
    int hd = t >> 6;
    float v = accum[i] / (denom[n * H + hd] + 1e-16f) + bias[t];
    if (relu) v = fmaxf(v, 0.f);
    out[i] = v;
}

// global mean pool (sums + counts)
__global__ void pool_kernel(const float* __restrict__ h3,
                            const int* __restrict__ batch,
                            float* __restrict__ pooled,
                            float* __restrict__ cnt) {
    int i = blockIdx.x * blockDim.x + threadIdx.x;   // over N*64
    if (i >= NNODES * 64) return;
    int n = i >> 6, c = i & 63;
    int g = batch[n];
    atomicAdd(&pooled[g * 64 + c], h3[i]);
    if (c == 0) atomicAdd(&cnt[g], 1.0f);
}

// out[g,d] = mean-pooled @ Wc + bc   (64x10, trivial)
__global__ void classify_kernel(const float* __restrict__ pooled,
                                const float* __restrict__ cnt,
                                const float* __restrict__ Wc,
                                const float* __restrict__ bc,
                                float* __restrict__ out) {
    int i = blockIdx.x * blockDim.x + threadIdx.x;   // over 64*10
    if (i >= NGRAPH * 10) return;
    int g = i / 10, d = i % 10;
    float inv = 1.0f / fmaxf(cnt[g], 1.0f);
    float s = bc[d];
    #pragma unroll 4
    for (int k = 0; k < 64; ++k) s += pooled[g * 64 + k] * inv * Wc[k * 10 + d];
    out[i] = s;
}

// =====================================================================
extern "C" void kernel_launch(void* const* d_in, const int* in_sizes, int n_in,
                              void* d_out, int out_size, void* d_ws, size_t ws_size,
                              hipStream_t stream) {
    (void)in_sizes; (void)n_in; (void)out_size; (void)ws_size;
    const float* x    = (const float*)d_in[0];
    const int*   ei   = (const int*)  d_in[1];
    const int*   bat  = (const int*)  d_in[2];
    const float* W1   = (const float*)d_in[3];
    const float* as1  = (const float*)d_in[4];
    const float* ad1  = (const float*)d_in[5];
    const float* b1   = (const float*)d_in[6];
    const float* W2   = (const float*)d_in[7];
    const float* as2  = (const float*)d_in[8];
    const float* ad2  = (const float*)d_in[9];
    const float* b2   = (const float*)d_in[10];
    const float* W3   = (const float*)d_in[11];
    const float* as3  = (const float*)d_in[12];
    const float* ad3  = (const float*)d_in[13];
    const float* b3   = (const float*)d_in[14];
    const float* Wc   = (const float*)d_in[15];
    const float* bc   = (const float*)d_in[16];
    float* out = (float*)d_out;

    // ---- workspace carve-out (floats) ----
    float* ws = (float*)d_ws;
    size_t off = 0;
    float* bufH   = ws + off; off += (size_t)NNODES * 256;  // GEMM output h
    float* bufA   = ws + off; off += (size_t)NNODES * 256;  // attn out L1 / L3
    float* bufB   = ws + off; off += (size_t)NNODES * 256;  // attn out L2
    float* alphaS = ws + off; off += (size_t)NNODES * 4;
    float* alphaD = ws + off; off += (size_t)NNODES * 4;
    float* emax   = ws + off; off += (size_t)NNODES * 4;
    float* denom  = ws + off; off += (size_t)NNODES * 4;
    float* eBuf   = ws + off; off += (size_t)ETOT   * 4;
    float* pooled = ws + off; off += NGRAPH * 64;           // pooled+cnt contiguous
    float* cnt    = ws + off; off += NGRAPH;

    const dim3 egrid((ETOT + 255) / 256);
    const dim3 agrid((ETOT + 3) / 4);
    const dim3 ggrid(NNODES / 16, 1);   // 2500 m-tiles; 8 waves x 2 n-tiles = 256 cols

    // ================= Layer 1 (H=4, Din=128 -> 256) =================
    gemm_wmma_f16<<<ggrid, 256, 0, stream>>>(x, W1, bufH, 128, 256);
    alpha_kernel<<<(NNODES * 4 + 255) / 256, 256, 0, stream>>>(bufH, as1, ad1, alphaS, alphaD, 4);
    init_node_kernel<<<(NNODES * 4 + 255) / 256, 256, 0, stream>>>(emax, denom, NNODES * 4);
    hipMemsetAsync(bufA, 0, (size_t)NNODES * 256 * sizeof(float), stream);
    edge_score_kernel<<<egrid, 256, 0, stream>>>(ei, alphaS, alphaD, eBuf, emax, 4);
    edge_exp_kernel  <<<egrid, 256, 0, stream>>>(ei, eBuf, emax, denom, 4);
    edge_agg_kernel  <<<agrid, 256, 0, stream>>>(ei, eBuf, bufH, bufA, 4);
    finalize_kernel<<<((size_t)NNODES * 256 + 255) / 256, 256, 0, stream>>>(bufA, denom, b1, bufA, 4, 1);

    // ================= Layer 2 (H=4, 256 -> 256) =====================
    gemm_wmma_f16<<<ggrid, 256, 0, stream>>>(bufA, W2, bufH, 256, 256);
    alpha_kernel<<<(NNODES * 4 + 255) / 256, 256, 0, stream>>>(bufH, as2, ad2, alphaS, alphaD, 4);
    init_node_kernel<<<(NNODES * 4 + 255) / 256, 256, 0, stream>>>(emax, denom, NNODES * 4);
    hipMemsetAsync(bufB, 0, (size_t)NNODES * 256 * sizeof(float), stream);
    edge_score_kernel<<<egrid, 256, 0, stream>>>(ei, alphaS, alphaD, eBuf, emax, 4);
    edge_exp_kernel  <<<egrid, 256, 0, stream>>>(ei, eBuf, emax, denom, 4);
    edge_agg_kernel  <<<agrid, 256, 0, stream>>>(ei, eBuf, bufH, bufB, 4);
    finalize_kernel<<<((size_t)NNODES * 256 + 255) / 256, 256, 0, stream>>>(bufB, denom, b2, bufB, 4, 1);

    // ================= Layer 3 (H=1, 256 -> 64) ======================
    gemm_wmma_f16<<<ggrid, 256, 0, stream>>>(bufB, W3, bufH, 256, 64);
    alpha_kernel<<<(NNODES + 255) / 256, 256, 0, stream>>>(bufH, as3, ad3, alphaS, alphaD, 1);
    init_node_kernel<<<(NNODES + 255) / 256, 256, 0, stream>>>(emax, denom, NNODES);
    hipMemsetAsync(bufA, 0, (size_t)NNODES * 64 * sizeof(float), stream);
    edge_score_kernel<<<egrid, 256, 0, stream>>>(ei, alphaS, alphaD, eBuf, emax, 1);
    edge_exp_kernel  <<<egrid, 256, 0, stream>>>(ei, eBuf, emax, denom, 1);
    edge_agg_kernel  <<<agrid, 64, 0, stream>>>(ei, eBuf, bufH, bufA, 1);
    finalize_kernel<<<((size_t)NNODES * 64 + 255) / 256, 256, 0, stream>>>(bufA, denom, b3, bufA, 1, 0);

    // ================= Mean pool + classifier ========================
    hipMemsetAsync(pooled, 0, (NGRAPH * 64 + NGRAPH) * sizeof(float), stream);
    pool_kernel<<<((size_t)NNODES * 64 + 255) / 256, 256, 0, stream>>>(bufA, bat, pooled, cnt);
    classify_kernel<<<10, 64, 0, stream>>>(pooled, cnt, Wc, bc, out);
}
